// NetWrapper_14207751815777
// MI455X (gfx1250) — matrix operand order repaired and checked by reading, not compiled, not run on hardware
//
#include <hip/hip_runtime.h>
#include <hip/hip_bf16.h>

// ---------------- problem constants (from reference) ----------------
#define BATCH 128
#define CIN   512
#define CH    512
#define HWPIX 784     // 28*28
#define HPIX  28
#define TT    64
#define FF    512
#define HM    224
#define KIDS  8
#define NUMNEG 64     // int(0.5*128)
#define THR_IDX 627   // int(784*0.8)

typedef __attribute__((ext_vector_type(16))) _Float16 v16h;
typedef __attribute__((ext_vector_type(8)))  float    v8f;
typedef __attribute__((ext_vector_type(4)))  unsigned v4u;
typedef __attribute__((ext_vector_type(8)))  int      v8i;
typedef __attribute__((ext_vector_type(4)))  int      v4i;

#define AS 34   // half stride (32 K + 2 pad) == TDM pad_interval=64B, pad_amount=4B
#define VS 516  // float stride (512 + 4 pad) for vis rows

#if defined(__gfx1250__) && __has_builtin(__builtin_amdgcn_tensor_load_to_lds)
#define USE_TDM 1
#else
#define USE_TDM 0
#endif

__device__ __forceinline__ v8f wmma16(v16h a, v16h b, v8f c) {
  // v_wmma_f32_16x16x32_f16: D = A(16x32) x B(32x16) + C
  return __builtin_amdgcn_wmma_f32_16x16x32_f16(false, a, false, b, (short)0, c,
                                                false, false);
}

// A fragment (16x32 f16, ISA 7.12.2): lane m holds row m; lanes<16 K={0..7,16..23},
// lanes>=16 K={8..15,24..31}; VGPR j holds a K-pair.
__device__ __forceinline__ v16h gatherA(const _Float16* aLds, int lane) {
  const int m  = lane & 15;
  const int kb = (lane >> 4) * 8;
  v16h a;
#pragma unroll
  for (int j = 0; j < 8; ++j) {
    const int k0 = ((j < 4) ? 2 * j : 16 + 2 * (j - 4)) + kb;
    union { unsigned u; _Float16 h[2]; } t;
    t.u = *reinterpret_cast<const unsigned*>(aLds + m * AS + k0);
    a[2 * j]     = t.h[0];
    a[2 * j + 1] = t.h[1];
  }
  return a;
}

// B fragment (32x16 f16): lane n holds column n; lanes<16 K={0..7,16..23},
// lanes>=16 K={8..15,24..31}; VGPR j holds a K-pair (same order as A).
__device__ __forceinline__ v16h gatherB(const _Float16* wLds, int lane, int nBase) {
  const int n  = (lane & 15) + nBase;
  const int kb = (lane >> 4) * 8;
  v16h b;
#pragma unroll
  for (int j = 0; j < 8; ++j) {
    const int k0 = ((j < 4) ? 2 * j : 16 + 2 * (j - 4)) + kb;
    union { unsigned u; _Float16 h[2]; } t;
    t.u = *reinterpret_cast<const unsigned*>(wLds + n * AS + k0);
    b[2 * j]     = t.h[0];
    b[2 * j + 1] = t.h[1];
  }
  return b;
}

#if USE_TDM
// raw LDS byte offset (addrspace(3) ptrtoint), as required by the D# lds_addr field
__device__ __forceinline__ unsigned ldsOffset(const void* p) {
  typedef __attribute__((address_space(3))) const unsigned char lds_byte;
  return (unsigned)(unsigned long long)(lds_byte*)(p);
}

// One TDM 2-D tile load: tile (tileD0 x tileD1) halfs from a row-major [d1][d0]
// f16 tensor with row stride stride0 (halfs). LDS layout gets 4B padding after
// every 64B stored -> effective row stride of 34 halfs (== AS).
__device__ __forceinline__ void tdm_load_2d(unsigned ldsAddr, const void* gptr,
                                            unsigned tileD0, unsigned tileD1,
                                            unsigned tensorD0, unsigned tensorD1,
                                            unsigned stride0) {
  const unsigned long long ga = (unsigned long long)(size_t)gptr;
  v4u g0;
  g0[0] = 1u;                                          // count=1, user mode
  g0[1] = ldsAddr;                                     // lds_addr [63:32]
  g0[2] = (unsigned)(ga & 0xffffffffu);                // global_addr [95:64]
  g0[3] = (unsigned)((ga >> 32) & 0x01ffffffu) | (2u << 30);  // addr hi + type=2
  v8i g1;
  g1[0] = (int)((1u << 16) | (1u << 20) | (3u << 22)); // data_size=2B, pad_en, pad_interval=64B, pad_amount=4B
  g1[1] = (int)((tensorD0 & 0xffffu) << 16);           // tensor_dim0 lo16 -> bits 63:48
  g1[2] = (int)(((tensorD0 >> 16) & 0xffffu) | ((tensorD1 & 0xffffu) << 16));
  g1[3] = (int)(((tensorD1 >> 16) & 0xffffu) | ((tileD0 & 0xffffu) << 16));
  g1[4] = (int)(tileD1 & 0xffffu);                     // tile_dim1 (tile_dim2=0)
  g1[5] = (int)stride0;                                // tensor_dim0_stride lo32
  g1[6] = 0;
  g1[7] = 0;
  const v4i gz4 = {0, 0, 0, 0};                        // groups 2/3 unused (2-D)
  const v8i gz8 = {0, 0, 0, 0, 0, 0, 0, 0};            // extra group (6-arg form)
  __builtin_amdgcn_tensor_load_to_lds(g0, g1, gz4, gz4, gz8, 0);
}

__device__ __forceinline__ void tdm_wait() {
#if __has_builtin(__builtin_amdgcn_s_wait_tensorcnt)
  __builtin_amdgcn_s_wait_tensorcnt(0);
#else
  asm volatile("s_wait_tensorcnt 0x0" ::: "memory");
#endif
}
#endif  // USE_TDM

// ---------------- k0: zero the loss slot ----------------
__global__ void k0_init(float* out) { out[0] = 0.0f; }

// ---------------- kW: W_frame -> f16, transposed to [c][k] ----------------
__global__ __launch_bounds__(256) void kW_cvt(const float* __restrict__ Wf,
                                              _Float16* __restrict__ WfH) {
  const int c = blockIdx.x;
  for (int k = threadIdx.x; k < CIN; k += 256)
    WfH[(size_t)c * CIN + k] = (_Float16)Wf[(size_t)k * CH + c];
}

// ---------------- k1: audio heads ----------------
// a_orig = l2( (sum_t spect) @ W_orig / T ), a_tn = l2( (sum_t spect) @ W_trans / T )
__global__ __launch_bounds__(256) void k1_audio(const float* __restrict__ spect,
                                                const float* __restrict__ Wo,
                                                const float* __restrict__ Wt,
                                                float* __restrict__ a_orig,
                                                _Float16* __restrict__ a_tnH) {
  const int b = blockIdx.x, t = threadIdx.x;
  __shared__ float ss[FF];
  __shared__ float ao[CH], at[CH];
  __shared__ float red[256];
  __shared__ float invO, invT;

  for (int f = t; f < FF; f += 256) {
    const float* sp = spect + ((size_t)b * TT) * FF + f;
    float s = 0.f;
    for (int tt = 0; tt < TT; ++tt) s += sp[(size_t)tt * FF];
    ss[f] = s;
  }
  __syncthreads();
  for (int c = t; c < CH; c += 256) {
    float so = 0.f, st = 0.f;
    for (int f = 0; f < FF; ++f) {
      const float v = ss[f];
      so += v * Wo[(size_t)f * CH + c];
      st += v * Wt[(size_t)f * CH + c];
    }
    ao[c] = so * (1.0f / TT);
    at[c] = st * (1.0f / TT);
  }
  __syncthreads();
  float p1 = 0.f, p2 = 0.f;
  for (int c = t; c < CH; c += 256) { p1 += ao[c] * ao[c]; p2 += at[c] * at[c]; }
  red[t] = p1; __syncthreads();
  for (int s = 128; s > 0; s >>= 1) { if (t < s) red[t] += red[t + s]; __syncthreads(); }
  if (t == 0) invO = 1.0f / fmaxf(sqrtf(red[0]), 1e-12f);
  __syncthreads();
  red[t] = p2; __syncthreads();
  for (int s = 128; s > 0; s >>= 1) { if (t < s) red[t] += red[t + s]; __syncthreads(); }
  if (t == 0) invT = 1.0f / fmaxf(sqrtf(red[0]), 1e-12f);
  __syncthreads();
  for (int c = t; c < CH; c += 256) {
    a_orig[(size_t)b * CH + c] = ao[c] * invO;
    a_tnH [(size_t)b * CH + c] = (_Float16)(at[c] * invT);
  }
}

// ---------------- k2: false-negative mask ----------------
__global__ __launch_bounds__(128) void k2_neg(const float* __restrict__ a_orig,
                                              float* __restrict__ fn) {
  const int b = blockIdx.x, j = threadIdx.x;
  __shared__ float rowb[CH];
  __shared__ float sim[BATCH];
  for (int f = j; f < CH; f += 128) rowb[f] = a_orig[(size_t)b * CH + f];
  __syncthreads();
  float s = 0.f;
  for (int f = 0; f < CH; ++f) s += rowb[f] * a_orig[(size_t)j * CH + f];
  sim[j] = s;
  __syncthreads();
  const float v = sim[j];
  int r = 0;
  for (int k = 0; k < BATCH; ++k) {
    const float u = sim[k];
    r += (u < v) || (u == v && k < j);   // stable rank == argsort position
  }
  const int sel = (r < NUMNEG) || (j == b);
  fn[(size_t)b * BATCH + j] = sel ? 0.f : 1.f;
}

// ---------------- k3: WMMA visual projection + cross-audio similarity ----------------
// Block = 8 waves, one 16-row pixel tile of one batch sample.
// Phase 1: vis(16x512) = frame_tile(16x512) x W_frame(512x512)   [64 wmma]
// Phase 2: D(16x128)   = l2(vis) x a_tn^T                        [16 wmma]
// B operands staged into LDS by the Tensor Data Mover (1 tensor_load per K step).
__global__ __launch_bounds__(256) void k3_vis(const float* __restrict__ frame,
                                              const _Float16* __restrict__ WfH,
                                              const _Float16* __restrict__ a_tnH,
                                              float* __restrict__ D,
                                              float* __restrict__ simOut) {
  const int tileM = blockIdx.x;            // 0..48 (49*16 = 784)
  const int b     = blockIdx.y;            // 0..127
  const int t     = threadIdx.x;
  const int wave  = t >> 5;
  const int lane  = t & 31;
  const int pBase = tileM * 16;

  __shared__ __attribute__((aligned(16))) _Float16 aLds[16 * AS];
  __shared__ __attribute__((aligned(16))) _Float16 wLds[512 * AS];
  __shared__ __attribute__((aligned(16))) float    visLds[16 * VS];
  __shared__ __attribute__((aligned(16))) _Float16 tLds[128 * AS];
  __shared__ float invN[16];
  __shared__ float normP[16 * 16];

  // -------- phase 1 --------
  v8f acc0 = {}, acc1 = {}, acc2 = {}, acc3 = {};
  const int ntB0 = wave * 4 * 16;          // this wave's first N column
  for (int k0 = 0; k0 < CIN; k0 += 32) {
#if USE_TDM
    if (wave == 0)                          // 32x512 f16 tile, padded to [c][34]
      tdm_load_2d(ldsOffset(wLds), WfH + k0, 32, CH, CIN, CH, CIN);
#endif
    for (int i = t; i < 16 * 32; i += 256) {       // A tile: frame[b, k, p] (K-major)
      const int kk = i >> 4, m = i & 15;
      aLds[m * AS + kk] =
          (_Float16)frame[((size_t)b * CIN + (k0 + kk)) * HWPIX + pBase + m];
    }
#if USE_TDM
    if (wave == 0) tdm_wait();
#else
    for (int i = t; i < 32 * CH; i += 256) {       // fallback: manual copy
      const int c = i & 511, kk = i >> 9;
      wLds[c * AS + kk] = WfH[(size_t)c * CIN + k0 + kk];
    }
#endif
    __syncthreads();
    const v16h af = gatherA(aLds, lane);
    acc0 = wmma16(af, gatherB(wLds, lane, ntB0),      acc0);
    acc1 = wmma16(af, gatherB(wLds, lane, ntB0 + 16), acc1);
    acc2 = wmma16(af, gatherB(wLds, lane, ntB0 + 32), acc2);
    acc3 = wmma16(af, gatherB(wLds, lane, ntB0 + 48), acc3);
    __syncthreads();
  }
  {                                         // spill vis tile: C layout, M = r + 8*(lane>=16)
    const int mBase = (lane >> 4) * 8;
    const int n = lane & 15;
#pragma unroll
    for (int r = 0; r < 8; ++r) {
      const int m = mBase + r;
      visLds[m * VS + ntB0 +      n] = acc0[r];
      visLds[m * VS + ntB0 + 16 + n] = acc1[r];
      visLds[m * VS + ntB0 + 32 + n] = acc2[r];
      visLds[m * VS + ntB0 + 48 + n] = acc3[r];
    }
  }
  __syncthreads();
  {                                         // per-pixel inverse L2 norm over channels
    const int r = t >> 4, c0 = t & 15;
    float s = 0.f;
    for (int c = c0; c < CH; c += 16) { const float v = visLds[r * VS + c]; s += v * v; }
    normP[r * 16 + c0] = s;
  }
  __syncthreads();
  if (t < 16) {
    float s = 0.f;
    for (int i = 0; i < 16; ++i) s += normP[t * 16 + i];
    invN[t] = 1.0f / fmaxf(sqrtf(s), 1e-12f);
  }
  __syncthreads();

  // -------- phase 2 --------
  v8f c2 = {};
  const int ntD = wave * 16;                // audio column tile for this wave
  for (int k0 = 0; k0 < CH; k0 += 32) {
#if USE_TDM
    if (wave == 0)                          // 32x128 f16 tile of a_tn^T
      tdm_load_2d(ldsOffset(tLds), a_tnH + k0, 32, BATCH, CH, BATCH, CH);
#endif
    v16h af;
    {
      const int m = lane & 15, kb = (lane >> 4) * 8;
      const float inv = invN[m];
#pragma unroll
      for (int j = 0; j < 8; ++j) {
        const int kq = ((j < 4) ? 2 * j : 16 + 2 * (j - 4)) + kb + k0;
        af[2 * j]     = (_Float16)(visLds[m * VS + kq]     * inv);
        af[2 * j + 1] = (_Float16)(visLds[m * VS + kq + 1] * inv);
      }
    }
#if USE_TDM
    if (wave == 0) tdm_wait();
#else
    for (int i = t; i < BATCH * 32; i += 256) {
      const int n = i >> 5, kk = i & 31;
      tLds[n * AS + kk] = a_tnH[(size_t)n * CH + k0 + kk];
    }
#endif
    __syncthreads();
    c2 = wmma16(af, gatherB(tLds, lane, ntD), c2);
    __syncthreads();
  }
  {
    const int mBase = (lane >> 4) * 8;
    const int dcol  = ntD + (lane & 15);
#pragma unroll
    for (int r = 0; r < 8; ++r) {
      const int p = pBase + mBase + r;
      const float v = c2[r];
      D[((size_t)b * HWPIX + p) * BATCH + dcol] = v;
      if (dcol == b) simOut[(size_t)b * HWPIX + p] = v;   // att map vs own audio
    }
  }
}

// ---------------- k4: binarize + mask logic -> pos1/pos2 ----------------
__global__ __launch_bounds__(256) void k4_pos(const float* __restrict__ sim1,
                                              const float* __restrict__ sim2,
                                              const int* __restrict__ mask1,
                                              const int* __restrict__ mask2,
                                              float* __restrict__ pos1,
                                              float* __restrict__ pos2) {
  const int b = blockIdx.x, t = threadIdx.x;
  __shared__ float s1[HWPIX], s2[HWPIX];
  __shared__ float bs1[HWPIX], bs2[HWPIX];
  __shared__ int   mm1[HWPIX], mm2[HWPIX];
  __shared__ float thr1, thr2;
  __shared__ int   pres1[KIDS], pres2[KIDS], scores[KIDS];
  __shared__ int   nz1, nz2, sumSm2;
  __shared__ int   chosen, emptyF, noShared;

  if (t < KIDS) { pres1[t] = 0; pres2[t] = 0; scores[t] = 0; }
  if (t == 0) { nz1 = 0; nz2 = 0; sumSm2 = 0; }
  for (int p = t; p < HWPIX; p += 256) {
    s1[p] = sim1[(size_t)b * HWPIX + p];
    s2[p] = sim2[(size_t)b * HWPIX + p];
  }
  __syncthreads();
  for (int p = t; p < HWPIX; p += 256) {            // nearest resize 224 -> 28 (stride 8)
    const int h = p / HPIX, w = p - h * HPIX;
    const int v1 = mask1[((size_t)b * HM + h * 8) * HM + w * 8];
    const int v2 = mask2[((size_t)b * HM + h * 8) * HM + w * 8];
    mm1[p] = v1; mm2[p] = v2;
    pres1[v1] = 1; pres2[v2] = 1;                   // benign same-value races
    if (v1 != 0) nz1 = 1;
    if (v2 != 0) nz2 = 1;
  }
  __syncthreads();
  for (int p = t; p < HWPIX; p += 256) {            // order statistic (stable rank)
    float v = s1[p]; int r = 0;
    for (int q = 0; q < HWPIX; ++q) { const float u = s1[q]; r += (u < v) || (u == v && q < p); }
    if (r == THR_IDX) thr1 = v;
    v = s2[p]; r = 0;
    for (int q = 0; q < HWPIX; ++q) { const float u = s2[q]; r += (u < v) || (u == v && q < p); }
    if (r == THR_IDX) thr2 = v;
  }
  __syncthreads();
  for (int p = t; p < HWPIX; p += 256) {            // minmax-norm is monotone -> compare raw
    bs1[p] = (s1[p] > thr1) ? 1.f : 0.f;
    bs2[p] = (s2[p] > thr2) ? 1.f : 0.f;
    if (bs1[p] > 0.f) atomicAdd(&scores[mm1[p]], 1);
  }
  __syncthreads();
  if (t == 0) {
    const int empty = (nz1 == 0) || (nz2 == 0);
    int anyShared = 0, ch = 0;
    float bestS = -3e9f;
    for (int id = 0; id < KIDS; ++id) {
      const int sh = pres1[id] && pres2[id];
      anyShared |= sh;
      const float sc = sh ? (float)scores[id] : -1e9f;
      if (sc > bestS) { bestS = sc; ch = id; }      // first max wins, like jnp.argmax
    }
    chosen = ch; emptyF = empty;
    noShared = (!anyShared) && (!empty);
  }
  __syncthreads();
  for (int p = t; p < HWPIX; p += 256) {
    const float bc1 = emptyF ? 1.f : (mm1[p] == chosen ? 1.f : 0.f);
    const float bc2 = emptyF ? 1.f : (mm2[p] == chosen ? 1.f : 0.f);
    const float sm2 = bs2[p] * bc2;
    if (sm2 > 0.f) atomicAdd(&sumSm2, 1);
    pos1[(size_t)b * HWPIX + p] = noShared ? bs1[p] : bs1[p] * bc1;
    s1[p] = bc2; s2[p] = sm2;                       // stash for second pass
  }
  __syncthreads();
  for (int p = t; p < HWPIX; p += 256) {
    const float p2 = (sumSm2 == 0) ? s1[p] : s2[p];
    pos2[(size_t)b * HWPIX + p] = noShared ? bs2[p] : p2;
  }
}

// ---------------- k5: masked max, log-softmax, loss ----------------
__global__ __launch_bounds__(128) void k5_loss(const float* __restrict__ D1,
                                               const float* __restrict__ D2,
                                               const float* __restrict__ pos1,
                                               const float* __restrict__ pos2,
                                               const float* __restrict__ fn,
                                               float* __restrict__ out) {
  const int b = blockIdx.x, d = threadIdx.x;
  __shared__ float P[HWPIX];
  __shared__ float red[BATCH];
  __shared__ float rowMax, lse, diag;
  const float fnv = fn[(size_t)b * BATCH + d];
  float lp[2];

#pragma unroll
  for (int view = 0; view < 2; ++view) {
    const float* Dv = (view == 0) ? D1 : D2;
    const float* Pv = (view == 0) ? pos1 : pos2;
    for (int p = d; p < HWPIX; p += 128) P[p] = Pv[(size_t)b * HWPIX + p];
    __syncthreads();
    const float* Db = Dv + (size_t)b * HWPIX * BATCH + d;
    float m = -1e30f;
    for (int p = 0; p < HWPIX; ++p) m = fmaxf(m, P[p] * Db[(size_t)p * BATCH]);
    if (fnv > 0.5f) m = -10.f;
    const float l = m * 5.0f;                       // / TEMP_D
    red[d] = l; __syncthreads();
    for (int s = 64; s > 0; s >>= 1) { if (d < s) red[d] = fmaxf(red[d], red[d + s]); __syncthreads(); }
    if (d == 0) rowMax = red[0];
    __syncthreads();
    red[d] = expf(l - rowMax); __syncthreads();
    for (int s = 64; s > 0; s >>= 1) { if (d < s) red[d] += red[d + s]; __syncthreads(); }
    if (d == 0) lse = logf(red[0]) + rowMax;
    if (d == b) diag = l;
    __syncthreads();
    lp[view] = diag - lse;
    __syncthreads();                                // before P is overwritten
  }
  if (d == 0) atomicAdd(out, -(lp[0] + lp[1]) * (1.0f / BATCH));
}

// ---------------- launcher ----------------
extern "C" void kernel_launch(void* const* d_in, const int* in_sizes, int n_in,
                              void* d_out, int out_size, void* d_ws, size_t ws_size,
                              hipStream_t stream) {
  const float* frame1 = (const float*)d_in[0];
  const float* frame2 = (const float*)d_in[1];
  const int*   mask1  = (const int*)d_in[2];
  const int*   mask2  = (const int*)d_in[3];
  const float* spect  = (const float*)d_in[4];
  const float* Wf     = (const float*)d_in[5];
  const float* Wo     = (const float*)d_in[6];
  const float* Wt     = (const float*)d_in[7];
  float* out = (float*)d_out;                       // [0]=loss, [1..]=sim1 (B,28,28)

  float* ws = (float*)d_ws;
  size_t off = 0;
  float*    a_orig = ws + off;  off += (size_t)BATCH * CH;
  float*    fn     = ws + off;  off += (size_t)BATCH * BATCH;
  float*    sim2   = ws + off;  off += (size_t)BATCH * HWPIX;
  float*    pos1   = ws + off;  off += (size_t)BATCH * HWPIX;
  float*    pos2   = ws + off;  off += (size_t)BATCH * HWPIX;
  _Float16* a_tnH  = (_Float16*)(ws + off);  off += (size_t)BATCH * CH / 2;
  _Float16* WfH    = (_Float16*)(ws + off);  off += (size_t)CIN * CH / 2;
  float*    D1     = ws + off;  off += (size_t)BATCH * HWPIX * BATCH;
  float*    D2     = ws + off;

  k0_init<<<1, 1, 0, stream>>>(out);
  kW_cvt<<<CH, 256, 0, stream>>>(Wf, WfH);
  k1_audio<<<BATCH, 256, 0, stream>>>(spect, Wo, Wt, a_orig, a_tnH);
  k2_neg<<<BATCH, 128, 0, stream>>>(a_orig, fn);
  k3_vis<<<dim3(HWPIX / 16, BATCH), 256, 0, stream>>>(frame1, WfH, a_tnH, D1, out + 1);
  k3_vis<<<dim3(HWPIX / 16, BATCH), 256, 0, stream>>>(frame2, WfH, a_tnH, D2, sim2);
  k4_pos<<<BATCH, 256, 0, stream>>>(out + 1, sim2, mask1, mask2, pos1, pos2);
  k5_loss<<<BATCH, 128, 0, stream>>>(D1, D2, pos1, pos2, fn, out);
}
